// Agent_34565896798870
// MI455X (gfx1250) — compile-verified, matrix operand order checked by reading, not run on hardware
//
#include <hip/hip_runtime.h>
#include <hip/hip_bf16.h>

typedef __attribute__((ext_vector_type(16))) _Float16 v16h;
typedef __attribute__((ext_vector_type(8)))  _Float16 v8h;
typedef __attribute__((ext_vector_type(8)))  float    v8f;
typedef __attribute__((ext_vector_type(4)))  int      v4i;

#define TSTEPS 128
#define BATCH  16
#define UNITS  128
#define NORM_SCALE 11.3137084989847604f  // sqrt(128)

// ---------------------------------------------------------------------------
// Async global->LDS copy (gfx1250 GLOBAL_LOAD_ASYNC_TO_LDS_B128, ASYNCcnt
// tracked) with a synchronous fallback if the builtin is unavailable.
// Builtin signature (probe-confirmed): (int4 AS1* src, int4 AS3* dst, Ii, Ii)
// ---------------------------------------------------------------------------
#if __has_builtin(__builtin_amdgcn_global_load_async_to_lds_b128)
#define HAS_ASYNC_LDS 1
#else
#define HAS_ASYNC_LDS 0
#endif

__device__ __forceinline__ void copy16_g2l(const void* g, void* l) {
#if HAS_ASYNC_LDS
  __builtin_amdgcn_global_load_async_to_lds_b128(
      (__attribute__((address_space(1))) v4i*)g,
      (__attribute__((address_space(3))) v4i*)l, 0, 0);
#else
  *(v8h*)l = *(const v8h*)g;
#endif
}

#if HAS_ASYNC_LDS
#define WAIT_ASYNC(n) asm volatile("s_wait_asynccnt " #n ::: "memory")
#else
#define WAIT_ASYNC(n) do {} while (0)
#endif

// ---------------------------------------------------------------------------
// WMMA helper: D = A(16x32 f16) * B(32x16 f16) + C(16x16 f32)
// ---------------------------------------------------------------------------
__device__ __forceinline__ v8f wmma_f16(v16h a, v16h b, v8f c) {
  return __builtin_amdgcn_wmma_f32_16x16x32_f16(
      /*neg_a=*/false, a, /*neg_b=*/false, b,
      /*c_mod=*/(short)0, c, /*reuse_a=*/false, /*reuse_b=*/false);
}

// Fragment loader for A (row-major MxK) or B (column-major: N rows of K).
// 16-bit operand striping: lane holds row (lane&15); lanes<16 take K j=0..7 and
// 16..23 of each 32-chunk, lanes>=16 take K j=8..15 and 24..31.
// Requires base 16B-aligned, ld multiple of 8, kb multiple of 32.
__device__ __forceinline__ v16h load_frag_k(const _Float16* base, int ld, int kb, int lane) {
  int r = lane & 15, hi = (lane >> 4) & 1;
  const _Float16* p = base + (size_t)r * ld + kb + hi * 8;
  v8h lo = *(const v8h*)p;
  v8h hv = *(const v8h*)(p + 16);
  v16h f;
#pragma unroll
  for (int j = 0; j < 8; ++j) { f[j] = lo[j]; f[8 + j] = hv[j]; }
  return f;
}

// ---------------------------------------------------------------------------
// Weight conversion kernels (one-shot, memory-light)
// ---------------------------------------------------------------------------
// OIHW f32 -> f16 [Co][KH*KW*Ci], k ordered (ky,kx,c) to match NHWC im2col
__global__ void conv_w_to_f16(const float* __restrict__ w, _Float16* __restrict__ wt,
                              int Co, int Ci, int KH, int KW) {
  int K = KH * KW * Ci, KWC = KW * Ci, total = Co * K;
  for (int idx = blockIdx.x * blockDim.x + threadIdx.x; idx < total;
       idx += gridDim.x * blockDim.x) {
    int o = idx / K, k = idx - o * K;
    int ky = k / KWC, rem = k - ky * KWC;
    int kx = rem / Ci, c = rem - kx * Ci;
    wt[idx] = (_Float16)w[((o * Ci + c) * KH + ky) * KW + kx];
  }
}

// fc_w [3136,512] (NCHW-flat k) -> f16 [512][3136] with k permuted to NHWC-flat
__global__ void fc_w_to_f16(const float* __restrict__ w, _Float16* __restrict__ wt) {
  int total = 512 * 3136;
  for (int idx = blockIdx.x * blockDim.x + threadIdx.x; idx < total;
       idx += gridDim.x * blockDim.x) {
    int o = idx / 3136, k = idx - o * 3136;      // k = (y*7+x)*64 + c  (NHWC)
    int yx = k >> 6, c = k & 63;
    wt[idx] = (_Float16)w[(c * 49 + yx) * 512 + o];  // ref k = c*49 + y*7 + x
  }
}

// [K,N] f32 row-major -> f16 col-major wt[n*K + k]
__global__ void mat_to_f16_colmajor(const float* __restrict__ w, _Float16* __restrict__ wt,
                                    int K, int N) {
  int total = K * N;
  for (int idx = blockIdx.x * blockDim.x + threadIdx.x; idx < total;
       idx += gridDim.x * blockDim.x) {
    int n = idx / K, k = idx - n * K;
    wt[idx] = (_Float16)w[k * N + n];
  }
}

// fused head weights: cols 0..5 = actor_w, col 6 = critic_w, 7..15 = 0; colmajor [16][256]
__global__ void build_head_w(const float* __restrict__ aw, const float* __restrict__ cw,
                             _Float16* __restrict__ wt) {
  int total = 16 * 256;
  for (int idx = blockIdx.x * blockDim.x + threadIdx.x; idx < total;
       idx += gridDim.x * blockDim.x) {
    int n = idx >> 8, k = idx & 255;
    float v = (n < 6) ? aw[k * 6 + n] : (n == 6 ? cw[k] : 0.0f);
    wt[idx] = (_Float16)v;
  }
}

// ---------------------------------------------------------------------------
// Implicit-im2col conv as WMMA GEMM.
// M = N_img*OH*OW patches, N = Co, K = KH*KW*Ci (k ordered (ky,kx,c)).
// The whole Co x K weight panel is staged once into LDS per block via async
// b128 copies; all k-iterations then read B fragments with ds_load_b128.
// Block = 256 threads = 8 waves; each wave owns one 16-row M-tile, all Co tiles.
// ---------------------------------------------------------------------------
template <bool IN_F32, int H, int W, int Ci, int KH, int KW, int S, int OH, int OW, int Co>
__global__ void conv_wmma_kernel(const void* __restrict__ xin,
                                 const _Float16* __restrict__ wt,
                                 const float* __restrict__ bias,
                                 _Float16* __restrict__ y) {
  constexpr int K = KH * KW * Ci;
  constexpr int KWC = KW * Ci;      // multiple of 8 for all three convs
  constexpr int OHOW = OH * OW;
  constexpr int NT = Co / 16;
  extern __shared__ __align__(16) char smem_raw[];
  _Float16* bsh = (_Float16*)smem_raw;              // [Co][K] col-major panel

  int wave = threadIdx.x >> 5, lane = threadIdx.x & 31;
  int r = lane & 15, hi = (lane >> 4) & 1;
  int mbase = (blockIdx.x * 8 + wave) * 16;

  // stage the weight panel into LDS (async, 16B per copy)
  constexpr int NCHUNK = (Co * K) / 8;
  for (int i = threadIdx.x; i < NCHUNK; i += 256)
    copy16_g2l(wt + (size_t)i * 8, bsh + (size_t)i * 8);

  // this lane's A-matrix row -> patch coordinates
  int m = mbase + r;
  int n = m / OHOW;
  int rem = m - n * OHOW;
  int oy = rem / OW, ox = rem - oy * OW;
  size_t rowoff0 = ((size_t)(n * H + oy * S) * W + ox * S) * Ci;

  WAIT_ASYNC(0);
  __syncthreads();

  v8f acc[NT] = {};
  for (int kb = 0; kb < K; kb += 32) {
    v16h a;
    int k0 = kb + hi * 8;
#pragma unroll
    for (int run = 0; run < 2; ++run) {
      int k = k0 + run * 16;
      int ky = k / KWC, kk = k - ky * KWC;   // 8-run never crosses a ky block (8 | KWC)
      size_t off = rowoff0 + (size_t)ky * W * Ci + kk;
      if constexpr (IN_F32) {
        const float* p = (const float*)xin + off;
#pragma unroll
        for (int j = 0; j < 8; ++j) a[run * 8 + j] = (_Float16)(p[j] * (1.0f / 255.0f));
      } else {
        v8h v = *(const v8h*)((const _Float16*)xin + off);
#pragma unroll
        for (int j = 0; j < 8; ++j) a[run * 8 + j] = v[j];
      }
    }
#pragma unroll
    for (int nt = 0; nt < NT; ++nt) {
      v16h b = load_frag_k(bsh + (size_t)nt * 16 * K, K, kb, lane);
      acc[nt] = wmma_f16(a, b, acc[nt]);
    }
  }
  // epilogue: C/D layout = lane col (lane&15), rows i + 8*(lane>=16)
#pragma unroll
  for (int nt = 0; nt < NT; ++nt) {
    int co = nt * 16 + r;
    float bv = bias[co];
#pragma unroll
    for (int i = 0; i < 8; ++i) {
      int mm = mbase + i + hi * 8;
      float v = fmaxf(acc[nt][i] + bv, 0.0f);
      y[(size_t)mm * Co + co] = (_Float16)v;
    }
  }
}

// ---------------------------------------------------------------------------
// Generic WMMA GEMM: C[M,N] = relu(A[M,K] * B(colmajor N,K) + bias)
// Block = 8 waves; wave -> one M-tile; blockIdx.y -> group of NT N-tiles.
// ---------------------------------------------------------------------------
template <int NT, bool RELU, bool HAS_BIAS, bool OUT_F32>
__global__ void gemm_wmma(const _Float16* __restrict__ A, const _Float16* __restrict__ Bw,
                          const float* __restrict__ bias, void* __restrict__ out,
                          int M, int N, int K) {
  int wave = threadIdx.x >> 5, lane = threadIdx.x & 31;
  int mbase = (blockIdx.x * 8 + wave) * 16;
  int nbase = blockIdx.y * NT * 16;
  const _Float16* Arow = A + (size_t)mbase * K;

  v8f acc[NT] = {};
  for (int kb = 0; kb < K; kb += 32) {
    if (kb + 32 < K)   // hint the A-panel stream (global_prefetch)
      __builtin_prefetch(Arow + (size_t)(lane & 15) * K + kb + 32, 0, 3);
    v16h a = load_frag_k(Arow, K, kb, lane);
#pragma unroll
    for (int nt = 0; nt < NT; ++nt) {
      v16h b = load_frag_k(Bw + (size_t)(nbase + nt * 16) * K, K, kb, lane);
      acc[nt] = wmma_f16(a, b, acc[nt]);
    }
  }
  int r = lane & 15, hi = (lane >> 4) & 1;
#pragma unroll
  for (int nt = 0; nt < NT; ++nt) {
    int col = nbase + nt * 16 + r;
    float bv = 0.0f;
    if constexpr (HAS_BIAS) bv = bias[col];
#pragma unroll
    for (int i = 0; i < 8; ++i) {
      int m = mbase + i + hi * 8;
      float v = acc[nt][i] + bv;
      if constexpr (RELU) v = fmaxf(v, 0.0f);
      if constexpr (OUT_F32) ((float*)out)[(size_t)m * N + col] = v;
      else ((_Float16*)out)[(size_t)m * N + col] = (_Float16)v;
    }
  }
}

// ---------------------------------------------------------------------------
// Unitary-RNN scan: single persistent workgroup (8 waves), T=128 steps.
// pre = st @ wrec + proj[t]   (complex, [16,128] @ [128,128])
//   pre_r = st_r@Wr + (-st_i)@Wi ; pre_i = st_r@Wi + st_i@Wr
// Each wave owns one 16-col N-tile; wrec B-fragments pre-loaded in registers.
// proj tiles for step t+1 are async-prefetched into double-buffered LDS while
// step t computes (hides global latency on the sequential critical path).
// Renorm each row to sqrt(U) via LDS f32 atomics.
// ---------------------------------------------------------------------------
__global__ void urnn_scan(const float* __restrict__ proj_r, const float* __restrict__ proj_i,
                          const float* __restrict__ done,
                          const float* __restrict__ s0r, const float* __restrict__ s0i,
                          const _Float16* __restrict__ wrec_r16,
                          const _Float16* __restrict__ wrec_i16,
                          _Float16* __restrict__ hreal) {
  __shared__ float sr32[BATCH * UNITS];
  __shared__ float si32[BATCH * UNITS];
  __shared__ float norm2[BATCH];
  __shared__ __align__(16) _Float16 ar16[BATCH * UNITS];
  __shared__ __align__(16) _Float16 ai16[BATCH * UNITS];
  __shared__ __align__(16) _Float16 ain16[BATCH * UNITS];
  __shared__ __align__(16) float pr_buf[2][BATCH * UNITS];   // proj double buffers
  __shared__ __align__(16) float pi_buf[2][BATCH * UNITS];

  int tid = threadIdx.x, lane = tid & 31, wave = tid >> 5;
  int r = lane & 15, hi = (lane >> 4) & 1;
  int ntile = wave;                       // 8 waves cover N = 128

  // issue: 4 async 16B copies per thread per step-group (2 for r, 2 for i)
  auto prefetch_proj = [&](int t, int buf) {
    const float* gr = proj_r + (size_t)t * BATCH * UNITS;
    const float* gi = proj_i + (size_t)t * BATCH * UNITS;
    for (int i = tid; i < (BATCH * UNITS) / 4; i += 256) {
      copy16_g2l(gr + i * 4, &pr_buf[buf][i * 4]);
      copy16_g2l(gi + i * 4, &pi_buf[buf][i * 4]);
    }
  };

  for (int idx = tid; idx < BATCH * UNITS; idx += 256) {
    sr32[idx] = s0r[idx];
    si32[idx] = s0i[idx];
  }
  prefetch_proj(0, 0);

  // wrec fragments held in registers for the whole scan (8 x v16h per wave)
  v16h br[4], bi[4];
#pragma unroll
  for (int kb = 0; kb < 4; ++kb) {
    br[kb] = load_frag_k(wrec_r16 + (size_t)ntile * 16 * UNITS, UNITS, kb * 32, lane);
    bi[kb] = load_frag_k(wrec_i16 + (size_t)ntile * 16 * UNITS, UNITS, kb * 32, lane);
  }
  WAIT_ASYNC(0);
  __syncthreads();

  for (int t = 0; t < TSTEPS; ++t) {
    int cur = t & 1;
    // kick off next step's proj tiles while we compute this one
    if (t + 1 < TSTEPS) prefetch_proj(t + 1, (t + 1) & 1);

    // reset-mask blend with init = 1 + 0i, convert to f16 A-operands
    for (int idx = tid; idx < BATCH * UNITS; idx += 256) {
      int b = idx >> 7;
      float d = done[t * BATCH + b];
      float str = (1.0f - d) * sr32[idx] + d * 1.0f;
      float sti = (1.0f - d) * si32[idx];
      ar16[idx]  = (_Float16)str;
      ai16[idx]  = (_Float16)sti;
      ain16[idx] = (_Float16)(-sti);
    }
    if (tid < BATCH) norm2[tid] = 0.0f;

    // group for step t must have landed; the newest 4 ops (t+1) may stay in flight
    if (t + 1 < TSTEPS) { WAIT_ASYNC(4); } else { WAIT_ASYNC(0); }
    __syncthreads();

    v8f accr = {}, acci = {};
#pragma unroll
    for (int kb = 0; kb < 4; ++kb) {
      v16h arf  = load_frag_k(ar16, UNITS, kb * 32, lane);
      v16h aif  = load_frag_k(ai16, UNITS, kb * 32, lane);
      v16h ainf = load_frag_k(ain16, UNITS, kb * 32, lane);
      accr = wmma_f16(arf,  br[kb], accr);
      accr = wmma_f16(ainf, bi[kb], accr);
      acci = wmma_f16(arf,  bi[kb], acci);
      acci = wmma_f16(aif,  br[kb], acci);
    }

    int col = ntile * 16 + r;
    float vr[8], vi[8];
#pragma unroll
    for (int i = 0; i < 8; ++i) {
      int b = i + hi * 8;
      int off = b * UNITS + col;
      vr[i] = accr[i] + pr_buf[cur][off];
      vi[i] = acci[i] + pi_buf[cur][off];
      atomicAdd(&norm2[b], vr[i] * vr[i] + vi[i] * vi[i]);
    }
    __syncthreads();

#pragma unroll
    for (int i = 0; i < 8; ++i) {
      int b = i + hi * 8;
      float scale = NORM_SCALE * rsqrtf(norm2[b] + 1e-30f);
      float nr = vr[i] * scale, ni = vi[i] * scale;
      sr32[b * UNITS + col] = nr;
      si32[b * UNITS + col] = ni;
      size_t m = (size_t)(t * BATCH + b);
      hreal[m * 256 + col]        = (_Float16)nr;   // real half
      hreal[m * 256 + 128 + col]  = (_Float16)ni;   // imag half
    }
    __syncthreads();
  }
}

// ---------------------------------------------------------------------------
// Fused actor/critic head: out[2048,7] = hreal[2048,256] @ Whead[256,16-pad]
// ---------------------------------------------------------------------------
__global__ void head_gemm(const _Float16* __restrict__ hreal, const _Float16* __restrict__ wh,
                          const float* __restrict__ actor_b, const float* __restrict__ critic_b,
                          float* __restrict__ out) {
  int wave = threadIdx.x >> 5, lane = threadIdx.x & 31;
  int mbase = (blockIdx.x * 8 + wave) * 16;
  v8f acc = {};
  for (int kb = 0; kb < 256; kb += 32) {
    v16h a = load_frag_k(hreal + (size_t)mbase * 256, 256, kb, lane);
    v16h b = load_frag_k(wh, 256, kb, lane);
    acc = wmma_f16(a, b, acc);
  }
  int r = lane & 15, hi = (lane >> 4) & 1;
  if (r < 7) {
    float bv = (r < 6) ? actor_b[r] : critic_b[0];
#pragma unroll
    for (int i = 0; i < 8; ++i) {
      int m = mbase + i + hi * 8;
      out[(size_t)m * 7 + r] = acc[i] + bv;
    }
  }
}

// ---------------------------------------------------------------------------
extern "C" void kernel_launch(void* const* d_in, const int* in_sizes, int n_in,
                              void* d_out, int out_size, void* d_ws, size_t ws_size,
                              hipStream_t stream) {
  (void)in_sizes; (void)n_in; (void)out_size; (void)ws_size;
  const float* x        = (const float*)d_in[0];
  const float* done     = (const float*)d_in[1];
  const float* s0r      = (const float*)d_in[2];
  const float* s0i      = (const float*)d_in[3];
  const float* conv1_w  = (const float*)d_in[4];
  const float* conv1_b  = (const float*)d_in[5];
  const float* conv2_w  = (const float*)d_in[6];
  const float* conv2_b  = (const float*)d_in[7];
  const float* conv3_w  = (const float*)d_in[8];
  const float* conv3_b  = (const float*)d_in[9];
  const float* fc_w     = (const float*)d_in[10];
  const float* fc_b     = (const float*)d_in[11];
  const float* win_r    = (const float*)d_in[12];
  const float* win_i    = (const float*)d_in[13];
  const float* wrec_r   = (const float*)d_in[14];
  const float* wrec_i   = (const float*)d_in[15];
  const float* actor_w  = (const float*)d_in[16];
  const float* actor_b  = (const float*)d_in[17];
  const float* critic_w = (const float*)d_in[18];
  const float* critic_b = (const float*)d_in[19];
  float* out = (float*)d_out;

  size_t off = 0;
  auto take = [&](size_t bytes) -> void* {
    void* p = (char*)d_ws + off;
    off += (bytes + 255) & ~(size_t)255;
    return p;
  };
  _Float16* wt1     = (_Float16*)take((size_t)32 * 192 * 2);
  _Float16* wt2     = (_Float16*)take((size_t)64 * 512 * 2);
  _Float16* wt3     = (_Float16*)take((size_t)64 * 576 * 2);
  _Float16* wtfc    = (_Float16*)take((size_t)512 * 3136 * 2);
  _Float16* wtwin_r = (_Float16*)take((size_t)128 * 512 * 2);
  _Float16* wtwin_i = (_Float16*)take((size_t)128 * 512 * 2);
  _Float16* wtrec_r = (_Float16*)take((size_t)128 * 128 * 2);
  _Float16* wtrec_i = (_Float16*)take((size_t)128 * 128 * 2);
  _Float16* wthead  = (_Float16*)take((size_t)16 * 256 * 2);
  _Float16* h1      = (_Float16*)take((size_t)2048 * 20 * 20 * 32 * 2);  // 52.4 MB
  _Float16* h2      = (_Float16*)take((size_t)2048 * 9 * 9 * 64 * 2);    // 21.2 MB
  _Float16* h3      = (_Float16*)take((size_t)2048 * 3136 * 2);          // 12.8 MB
  _Float16* hfc     = (_Float16*)take((size_t)2048 * 512 * 2);           //  2.1 MB
  float*    projr   = (float*)take((size_t)2048 * 128 * 4);
  float*    proji   = (float*)take((size_t)2048 * 128 * 4);
  _Float16* hreal   = (_Float16*)take((size_t)2048 * 256 * 2);

  // --- one-shot weight conversions ---
  conv_w_to_f16<<<24, 256, 0, stream>>>(conv1_w, wt1, 32, 3, 8, 8);
  conv_w_to_f16<<<128, 256, 0, stream>>>(conv2_w, wt2, 64, 32, 4, 4);
  conv_w_to_f16<<<144, 256, 0, stream>>>(conv3_w, wt3, 64, 64, 3, 3);
  fc_w_to_f16<<<3136, 256, 0, stream>>>(fc_w, wtfc);
  mat_to_f16_colmajor<<<256, 256, 0, stream>>>(win_r, wtwin_r, 512, 128);
  mat_to_f16_colmajor<<<256, 256, 0, stream>>>(win_i, wtwin_i, 512, 128);
  mat_to_f16_colmajor<<<64, 256, 0, stream>>>(wrec_r, wtrec_r, 128, 128);
  mat_to_f16_colmajor<<<64, 256, 0, stream>>>(wrec_i, wtrec_i, 128, 128);
  build_head_w<<<16, 256, 0, stream>>>(actor_w, critic_w, wthead);

  // --- CNN backbone (implicit-im2col WMMA, weight panel staged in LDS) ---
  // conv1: M = 2048*400 = 819200 -> 51200 tiles -> 6400 blocks; panel 12 KB
  conv_wmma_kernel<true, 84, 84, 3, 8, 8, 4, 20, 20, 32>
      <<<6400, 256, 32 * 192 * 2, stream>>>(x, wt1, conv1_b, h1);
  // conv2: M = 2048*81 = 165888 -> 1296 blocks; panel 64 KB
  conv_wmma_kernel<false, 20, 20, 32, 4, 4, 2, 9, 9, 64>
      <<<1296, 256, 64 * 512 * 2, stream>>>(h1, wt2, conv2_b, h2);
  // conv3: M = 2048*49 = 100352 -> 784 blocks; panel 72 KB
  conv_wmma_kernel<false, 9, 9, 64, 3, 3, 1, 7, 7, 64>
      <<<784, 256, 64 * 576 * 2, stream>>>(h2, wt3, conv3_b, h3);

  // FC: [2048,3136] @ [3136,512] + bias, relu -> f16
  gemm_wmma<4, true, true, false>
      <<<dim3(16, 8), 256, 0, stream>>>(h3, wtfc, fc_b, hfc, 2048, 512, 3136);

  // input projections: [2048,512] @ [512,128] -> f32 (consumed by scan)
  gemm_wmma<8, false, false, true>
      <<<dim3(16, 1), 256, 0, stream>>>(hfc, wtwin_r, nullptr, projr, 2048, 128, 512);
  gemm_wmma<8, false, false, true>
      <<<dim3(16, 1), 256, 0, stream>>>(hfc, wtwin_i, nullptr, proji, 2048, 128, 512);

  // sequential unitary-RNN scan: one persistent workgroup, wrec in registers,
  // proj tiles async double-buffered through LDS
  urnn_scan<<<1, 256, 0, stream>>>(projr, proji, done, s0r, s0i, wtrec_r, wtrec_i, hreal);

  // fused actor/critic head: [2048,256] @ [256,7] -> d_out
  head_gemm<<<16, 256, 0, stream>>>(hreal, wthead, actor_b, critic_b, out);
}
